// GPTBase_2783138808337
// MI455X (gfx1250) — compile-verified, matrix operand order checked by reading, not run on hardware
//
#include <hip/hip_runtime.h>

// ---------------------------------------------------------------- types
typedef long long i64;
typedef __bf16 v16bf __attribute__((ext_vector_type(16)));
typedef float  v8f   __attribute__((ext_vector_type(8)));

union Frag { uint4 q[2]; v16bf v; };
union U8   { uint4 q; unsigned short s[8]; };

__device__ __forceinline__ unsigned short f2bf(float f) {
  unsigned int u = __float_as_uint(f);
  unsigned int r = (u + 0x7FFFu + ((u >> 16) & 1u)) >> 16;
  return (unsigned short)r;
}
__device__ __forceinline__ float bf2f(unsigned short s) {
  return __uint_as_float(((unsigned int)s) << 16);
}

// block = 256 threads (8 waves) assumed for all reduction kernels
__device__ __forceinline__ float blockReduceSum(float v, float* red) {
  int lane = threadIdx.x & 31, w = threadIdx.x >> 5;
  for (int o = 16; o > 0; o >>= 1) v += __shfl_xor(v, o, 32);
  __syncthreads();
  if (lane == 0) red[w] = v;
  __syncthreads();
  float tot = 0.f;
#pragma unroll
  for (int k = 0; k < 8; k++) tot += red[k];
  return tot;
}
__device__ __forceinline__ float blockReduceMax(float v, float* red) {
  int lane = threadIdx.x & 31, w = threadIdx.x >> 5;
  for (int o = 16; o > 0; o >>= 1) v = fmaxf(v, __shfl_xor(v, o, 32));
  __syncthreads();
  if (lane == 0) red[w] = v;
  __syncthreads();
  float tot = -3.0e38f;
#pragma unroll
  for (int k = 0; k < 8; k++) tot = fmaxf(tot, red[k]);
  return tot;
}

// ---------------------------------------------------------------- WMMA GEMM
// C[M,N] = A[M,K](bf16) * B(bf16) (+C), block tile 64x128, K-step 32.
// flags: 1=accumulate into Cf, 2=GELU, 4=causal tile skip (n0>m0+63),
//        8=causal K limit (K <= m0+64). Cb!=null -> bf16 output, else Cf f32.
#define BM 64
#define BN 128
#define BK 32
#define LP 40  // LDS row pitch (ushort): 80B, 16B-aligned, bank-conflict free

__global__ __launch_bounds__(256) void wmma_gemm_kernel(
    const unsigned short* __restrict__ A, i64 lda, i64 strA,
    const unsigned short* __restrict__ Bm, i64 ldb, i64 strB, int transB,
    float* __restrict__ Cf, unsigned short* __restrict__ Cb, i64 ldc, i64 strC,
    int M, int N, int K, int flags)
{
  __shared__ unsigned short As[BM * LP];
  __shared__ unsigned short Bs[BN * LP];
  const int tid = threadIdx.x;
  const int m0 = blockIdx.y * BM;
  const int n0 = blockIdx.x * BN;
  const int z  = blockIdx.z;
  if ((flags & 4) && (n0 > m0 + BM - 1)) return;  // fully above diagonal
  A  += (i64)z * strA;
  Bm += (i64)z * strB;
  const int kEnd = (flags & 8) ? ((K < m0 + BM) ? K : (m0 + BM)) : K;

  const int lane = tid & 31;
  const int wave = tid >> 5;
  const int hi   = lane >> 4;     // K-half select
  const int lm   = lane & 15;     // row/col within 16
  const int wm   = (wave >> 2) * 32;
  const int wn   = (wave & 3) * 32;

  v8f acc[2][2];
#pragma unroll
  for (int i = 0; i < 2; i++)
#pragma unroll
    for (int j = 0; j < 2; j++)
#pragma unroll
      for (int e = 0; e < 8; e++) acc[i][j][e] = 0.f;

  const int arow = tid >> 2;
  const int akk  = (tid & 3) * 8;

  for (int k0 = 0; k0 < kEnd; k0 += BK) {
    // ---- stage A tile (64x32) : 1 uint4 per thread, coalesced
    {
      uint4 g = *(const uint4*)(A + (i64)(m0 + arow) * lda + (k0 + akk));
      *(uint4*)(&As[arow * LP + akk]) = g;
      if (k0 + BK < kEnd)
        __builtin_prefetch((const void*)(A + (i64)(m0 + arow) * lda + (k0 + BK + akk)), 0, 1);
    }
    // ---- stage B tile as Bs[n][k] (128x32)
    if (transB) {  // Bm is N x K row-major: direct copy
#pragma unroll
      for (int it = 0; it < 2; ++it) {
        int slot = tid + it * 256;
        int n = slot >> 2;
        int kk = (slot & 3) * 8;
        uint4 g = *(const uint4*)(Bm + (i64)(n0 + n) * ldb + (k0 + kk));
        *(uint4*)(&Bs[n * LP + kk]) = g;
      }
    } else {       // Bm is K x N row-major: transpose into LDS
#pragma unroll
      for (int it = 0; it < 2; ++it) {
        int slot = tid + it * 256;
        int k = slot >> 4;
        int n = (slot & 15) * 8;
        U8 g;
        g.q = *(const uint4*)(Bm + (i64)(k0 + k) * ldb + (n0 + n));
#pragma unroll
        for (int j = 0; j < 8; j++) Bs[(n + j) * LP + k] = g.s[j];
      }
    }
    __syncthreads();

    // ---- fragment gather (ISA 7.12.2 bf16 layout) + 4 WMMAs
    Frag a[2], b[2];
#pragma unroll
    for (int s = 0; s < 2; s++) {
      int r = wm + s * 16 + lm;
      a[s].q[0] = *(const uint4*)(&As[r * LP + hi * 8]);
      a[s].q[1] = *(const uint4*)(&As[r * LP + 16 + hi * 8]);
    }
#pragma unroll
    for (int s = 0; s < 2; s++) {
      int r = wn + s * 16 + lm;
      b[s].q[0] = *(const uint4*)(&Bs[r * LP + hi * 8]);
      b[s].q[1] = *(const uint4*)(&Bs[r * LP + 16 + hi * 8]);
    }
#pragma unroll
    for (int i = 0; i < 2; i++)
#pragma unroll
      for (int j = 0; j < 2; j++)
        acc[i][j] = __builtin_amdgcn_wmma_f32_16x16x32_bf16(
            false, a[i].v, false, b[j].v, (short)0, acc[i][j], false, false);
    __syncthreads();
  }

  // ---- epilogue
  float* cf = Cf ? Cf + (i64)z * strC : nullptr;
  unsigned short* cb = Cb ? Cb + (i64)z * strC : nullptr;
#pragma unroll
  for (int i = 0; i < 2; i++)
#pragma unroll
    for (int j = 0; j < 2; j++)
#pragma unroll
      for (int e = 0; e < 8; e++) {
        int m = m0 + wm + i * 16 + e + 8 * hi;
        int n = n0 + wn + j * 16 + lm;
        i64 off = (i64)m * ldc + n;
        float val = acc[i][j][e];
        if (flags & 2) val = 0.5f * val * (1.0f + erff(val * 0.70710678118f));
        if (cb) {
          cb[off] = f2bf(val);
        } else {
          if (flags & 1) val += cf[off];
          cf[off] = val;
        }
      }
}

// ---------------------------------------------------------------- small kernels
__global__ void embed_kernel(const int* __restrict__ idx, const float* __restrict__ wte,
                             float* __restrict__ x, int C) {
  int row = blockIdx.x;
  int t = idx[row];
  for (int c = threadIdx.x; c < C; c += blockDim.x)
    x[(i64)row * C + c] = wte[(i64)t * C + c];
}

__global__ void convert_kernel(const float* __restrict__ in, unsigned short* __restrict__ out, i64 n) {
  i64 i = (i64)blockIdx.x * blockDim.x + threadIdx.x;
  i64 st = (i64)gridDim.x * blockDim.x;
  for (; i < n; i += st) out[i] = f2bf(in[i]);
}

// row LayerNorm -> bf16 (C==1024, block 256)
__global__ __launch_bounds__(256) void ln_bf16_kernel(const float* __restrict__ x,
                                                      const float* __restrict__ w,
                                                      unsigned short* __restrict__ out, int C) {
  __shared__ float red[8];
  int row = blockIdx.x;
  const float* xr = x + (i64)row * C;
  float v[4]; float s = 0.f, ss = 0.f;
#pragma unroll
  for (int c = 0; c < 4; c++) {
    v[c] = xr[threadIdx.x + c * 256];
    s += v[c]; ss += v[c] * v[c];
  }
  s = blockReduceSum(s, red);
  __syncthreads();
  ss = blockReduceSum(ss, red);
  float mu = s / C;
  float inv = rsqrtf(ss / C - mu * mu + 1e-5f);
#pragma unroll
  for (int c = 0; c < 4; c++) {
    int cc = threadIdx.x + c * 256;
    out[(i64)row * C + cc] = f2bf((v[c] - mu) * inv * w[cc]);
  }
}

// x += LayerNorm(m) * w   (C==1024, block 256)
__global__ __launch_bounds__(256) void ln_add_kernel(const float* __restrict__ m,
                                                     const float* __restrict__ w,
                                                     float* __restrict__ x, int C) {
  __shared__ float red[8];
  int row = blockIdx.x;
  const float* mr = m + (i64)row * C;
  float v[4]; float s = 0.f, ss = 0.f;
#pragma unroll
  for (int c = 0; c < 4; c++) {
    v[c] = mr[threadIdx.x + c * 256];
    s += v[c]; ss += v[c] * v[c];
  }
  s = blockReduceSum(s, red);
  __syncthreads();
  ss = blockReduceSum(ss, red);
  float mu = s / C;
  float inv = rsqrtf(ss / C - mu * mu + 1e-5f);
#pragma unroll
  for (int c = 0; c < 4; c++) {
    int cc = threadIdx.x + c * 256;
    x[(i64)row * C + cc] += (v[c] - mu) * inv * w[cc];
  }
}

// att = softmax_j( scale*(att1[i,j] + qe[i, i-j]) ), causal, bf16 out. T==2048.
__global__ __launch_bounds__(256) void attn_softmax_kernel(
    const float* __restrict__ att1, const float* __restrict__ qe,
    unsigned short* __restrict__ p, int T, float scale) {
  __shared__ float red[8];
  int b = blockIdx.y, i = blockIdx.x, tid = threadIdx.x;
  i64 base = ((i64)b * T + i) * (i64)T;
  float v[8];
  float mx = -3.0e38f;
#pragma unroll
  for (int c = 0; c < 8; c++) {
    int j = tid + c * 256;
    float val = -3.0e38f;
    if (j <= i) val = scale * (att1[base + j] + qe[base + (i - j)]);
    v[c] = val;
    mx = fmaxf(mx, val);
  }
  mx = blockReduceMax(mx, red);
  __syncthreads();
  float sum = 0.f;
#pragma unroll
  for (int c = 0; c < 8; c++) {
    int j = tid + c * 256;
    float e = (j <= i) ? expf(v[c] - mx) : 0.f;
    v[c] = e;
    sum += e;
  }
  sum = blockReduceSum(sum, red);
  float inv = 1.f / sum;
#pragma unroll
  for (int c = 0; c < 8; c++) {
    int j = tid + c * 256;
    p[base + j] = f2bf(v[c] * inv);
  }
}

// s[b,i,r] = (r<=i) ? att[b,i,i-r] : 0
__global__ void gather_s_kernel(const unsigned short* __restrict__ p,
                                unsigned short* __restrict__ s, int T, i64 total) {
  i64 i = (i64)blockIdx.x * blockDim.x + threadIdx.x;
  i64 st = (i64)gridDim.x * blockDim.x;
  for (; i < total; i += st) {
    i64 row = i / T;            // b*T + qi
    int r = (int)(i - row * T);
    int qi = (int)(row % T);
    s[i] = (r <= qi) ? p[row * (i64)T + (qi - r)] : (unsigned short)0;
  }
}

// logits = x @ W(C x 2) + b ; per-row NLL. C==1024, block 256.
__global__ __launch_bounds__(256) void head_kernel(
    const float* __restrict__ x, const float* __restrict__ w, const float* __restrict__ bias,
    const int* __restrict__ tgt, float* __restrict__ logits, float* __restrict__ loss_rows, int C) {
  __shared__ float red[8];
  int row = blockIdx.x;
  const float* xr = x + (i64)row * C;
  float d0 = 0.f, d1 = 0.f;
#pragma unroll
  for (int c = 0; c < 4; c++) {
    int cc = threadIdx.x + c * 256;
    float xv = xr[cc];
    d0 += xv * w[cc * 2];
    d1 += xv * w[cc * 2 + 1];
  }
  d0 = blockReduceSum(d0, red);
  __syncthreads();
  d1 = blockReduceSum(d1, red);
  if (threadIdx.x == 0) {
    float l0 = d0 + bias[0], l1 = d1 + bias[1];
    logits[(i64)row * 2] = l0;
    logits[(i64)row * 2 + 1] = l1;
    float mx = fmaxf(l0, l1);
    float lse = mx + logf(expf(l0 - mx) + expf(l1 - mx));
    float lp = ((tgt[row] == 0) ? l0 : l1) - lse;
    loss_rows[row] = -lp;
  }
}

__global__ __launch_bounds__(256) void loss_reduce_kernel(const float* __restrict__ loss_rows,
                                                          float* __restrict__ out, int BT) {
  __shared__ float red[8];
  float s = 0.f;
  for (int i = threadIdx.x; i < BT; i += 256) s += loss_rows[i];
  s = blockReduceSum(s, red);
  if (threadIdx.x == 0) out[0] = s / (float)BT;
}

// ---------------------------------------------------------------- host
extern "C" void kernel_launch(void* const* d_in, const int* in_sizes, int n_in,
                              void* d_out, int out_size, void* d_ws, size_t ws_size,
                              hipStream_t stream) {
  constexpr int L = 6, Bc = 4, T = 2048, C = 1024;
  constexpr int BT = Bc * T;          // 8192
  constexpr int C3 = 3 * C, C4 = 4 * C;

  const int*   idx       = (const int*)d_in[0];
  const int*   targets   = (const int*)d_in[1];
  const float* wte       = (const float*)d_in[2];
  const float* ln1_w     = (const float*)d_in[3];
  const float* attn_w    = (const float*)d_in[4];
  const float* embk      = (const float*)d_in[5];
  const float* embv      = (const float*)d_in[6];
  const float* proj_w    = (const float*)d_in[7];
  const float* ln2_w     = (const float*)d_in[8];
  const float* fc_w      = (const float*)d_in[9];
  const float* fcp_w     = (const float*)d_in[10];
  const float* lm_w      = (const float*)d_in[11];
  const float* lm_b      = (const float*)d_in[12];
  float* out = (float*)d_out;

  size_t off = 0;
  auto alloc = [&](size_t bytes) -> void* {
    void* p = (char*)d_ws + off;
    off += (bytes + 255) & ~(size_t)255;
    return p;
  };
  unsigned short* wb_attn = (unsigned short*)alloc((size_t)L * C * C3 * 2);
  unsigned short* wb_proj = (unsigned short*)alloc((size_t)L * C * C * 2);
  unsigned short* wb_fc   = (unsigned short*)alloc((size_t)L * C * C4 * 2);
  unsigned short* wb_fcp  = (unsigned short*)alloc((size_t)L * C4 * C * 2);
  unsigned short* wb_ek   = (unsigned short*)alloc((size_t)L * T * C * 2);
  unsigned short* wb_ev   = (unsigned short*)alloc((size_t)L * T * C * 2);
  float*          x_f     = (float*)alloc((size_t)BT * C * 4);
  unsigned short* h_bf    = (unsigned short*)alloc((size_t)BT * C * 2);
  unsigned short* qkv_bf  = (unsigned short*)alloc((size_t)BT * C3 * 2);
  float*          att1_f  = (float*)alloc((size_t)Bc * T * T * 4);
  float*          qe_f    = (float*)alloc((size_t)Bc * T * T * 4);
  unsigned short* att_bf  = (unsigned short*)alloc((size_t)Bc * T * T * 2);
  unsigned short* s_bf    = (unsigned short*)alloc((size_t)Bc * T * T * 2);
  float*          y_f     = (float*)alloc((size_t)BT * C * 4);
  unsigned short* m_bf    = (unsigned short*)alloc((size_t)BT * C4 * 2);
  float*          loss_r  = (float*)alloc((size_t)BT * 4);

  // ---- one-time (per launch) bf16 weight conversion
  convert_kernel<<<2048, 256, 0, stream>>>(attn_w, wb_attn, (i64)L * C * C3);
  convert_kernel<<<2048, 256, 0, stream>>>(proj_w, wb_proj, (i64)L * C * C);
  convert_kernel<<<2048, 256, 0, stream>>>(fc_w,   wb_fc,   (i64)L * C * C4);
  convert_kernel<<<2048, 256, 0, stream>>>(fcp_w,  wb_fcp,  (i64)L * C4 * C);
  convert_kernel<<<2048, 256, 0, stream>>>(embk,   wb_ek,   (i64)L * T * C);
  convert_kernel<<<2048, 256, 0, stream>>>(embv,   wb_ev,   (i64)L * T * C);

  // ---- embedding
  embed_kernel<<<BT, 256, 0, stream>>>(idx, wte, x_f, C);

  const float scale = 1.0f / 32.0f;  // 1/sqrt(1024)
  for (int l = 0; l < L; l++) {
    // h = LN(x, ln1_w) -> bf16
    ln_bf16_kernel<<<BT, 256, 0, stream>>>(x_f, ln1_w + (i64)l * C, h_bf, C);

    // qkv = h @ attn_w[l]  (8192 x 3072, K=1024) -> bf16
    wmma_gemm_kernel<<<dim3(C3 / BN, BT / BM, 1), 256, 0, stream>>>(
        h_bf, C, 0, wb_attn + (i64)l * C * C3, C3, 0, 0,
        nullptr, qkv_bf, C3, 0, BT, C3, C, 0);

    // att1 = q @ k^T  (per-batch T x T, causal tile skip)
    wmma_gemm_kernel<<<dim3(T / BN, T / BM, Bc), 256, 0, stream>>>(
        qkv_bf, C3, (i64)T * C3, qkv_bf + C, C3, (i64)T * C3, 1,
        att1_f, nullptr, T, (i64)T * T, T, T, C, 4);

    // qe = q @ ek^T  (per-batch T x T, causal tile skip; ek shared)
    wmma_gemm_kernel<<<dim3(T / BN, T / BM, Bc), 256, 0, stream>>>(
        qkv_bf, C3, (i64)T * C3, wb_ek + (i64)l * T * C, C, 0, 1,
        qe_f, nullptr, T, (i64)T * T, T, T, C, 4);

    // causal softmax of scale*(att1 + qe gathered on rel) -> att_bf
    attn_softmax_kernel<<<dim3(T, Bc), 256, 0, stream>>>(att1_f, qe_f, att_bf, T, scale);

    // s[b,i,r] = att[b,i,i-r]
    gather_s_kernel<<<4096, 256, 0, stream>>>(att_bf, s_bf, T, (i64)Bc * T * T);

    // y = att @ v   (causal K-limit)
    wmma_gemm_kernel<<<dim3(C / BN, T / BM, Bc), 256, 0, stream>>>(
        att_bf, T, (i64)T * T, qkv_bf + 2 * C, C3, (i64)T * C3, 0,
        y_f, nullptr, C, (i64)T * C, T, C, T, 8);

    // y += s @ ev   (causal K-limit, accumulate)
    wmma_gemm_kernel<<<dim3(C / BN, T / BM, Bc), 256, 0, stream>>>(
        s_bf, T, (i64)T * T, wb_ev + (i64)l * T * C, C, 0, 0,
        y_f, nullptr, C, (i64)T * C, T, C, T, 8 | 1);

    // x += y @ proj_w[l]
    convert_kernel<<<2048, 256, 0, stream>>>(y_f, h_bf, (i64)BT * C);
    wmma_gemm_kernel<<<dim3(C / BN, BT / BM, 1), 256, 0, stream>>>(
        h_bf, C, 0, wb_proj + (i64)l * C * C, C, 0, 0,
        x_f, nullptr, C, 0, BT, C, C, 1);

    // m = gelu(x @ fc_w[l]) -> bf16
    convert_kernel<<<2048, 256, 0, stream>>>(x_f, h_bf, (i64)BT * C);
    wmma_gemm_kernel<<<dim3(C4 / BN, BT / BM, 1), 256, 0, stream>>>(
        h_bf, C, 0, wb_fc + (i64)l * C * C4, C4, 0, 0,
        nullptr, m_bf, C4, 0, BT, C4, C, 2);

    // mm = m @ fc_proj_w[l]
    wmma_gemm_kernel<<<dim3(C / BN, BT / BM, 1), 256, 0, stream>>>(
        m_bf, C4, 0, wb_fcp + (i64)l * C4 * C, C, 0, 0,
        y_f, nullptr, C, 0, BT, C, C4, 0);

    // x += LN(mm, ln2_w)
    ln_add_kernel<<<BT, 256, 0, stream>>>(y_f, ln2_w + (i64)l * C, x_f, C);
  }

  // logits + loss (deterministic two-stage reduction)
  head_kernel<<<BT, 256, 0, stream>>>(x_f, lm_w, lm_b, targets, out, loss_r, C);
  loss_reduce_kernel<<<1, 256, 0, stream>>>(loss_r, out + (i64)BT * 2, BT);
}
